// MLPPredictor_4836133175444
// MI455X (gfx1250) — matrix operand order repaired and checked by reading, not compiled, not run on hardware
//
#include <hip/hip_runtime.h>
#include <hip/hip_bf16.h>

typedef __attribute__((ext_vector_type(8)))  _Float16 v8h;
typedef __attribute__((ext_vector_type(16))) _Float16 v16h;
typedef __attribute__((ext_vector_type(8)))  float    v8f;

#define NN_NODES 50000
#define NN_EDGES 500000
#define HF       512
#define KDIM     1024
#define M_BLK    128
#define THREADS  512
#define ROWH     40            // padded LDS row stride in halves (20 dwords)

union V16u { v16h v; v8h h[2]; };

// Prep 1: pack W1 [1024][512] f32 into WMMA-B fragment order, f16:
//   BP[ntile][kchunk][lane][j] = W1[kchunk*32 + (lane>>4)*16 + j][ntile*16 + (lane&15)]
// so each wave's B tile load is one contiguous 1 KB burst (32 lanes x 32 B).
__global__ void w1_pack_f16(const float* __restrict__ W1,
                            _Float16* __restrict__ BP) {
    int tid = blockIdx.x * blockDim.x + threadIdx.x;   // 0 .. 512*1024-1
    int j     = tid & 15;
    int lane  = (tid >> 4) & 31;
    int kc    = (tid >> 9) & 31;
    int ntile = tid >> 14;
    int k = kc * 32 + (lane >> 4) * 16 + j;
    int n = ntile * 16 + (lane & 15);
    BP[tid] = (_Float16)W1[k * HF + n];
}

// Prep 2: h f32 -> f16 (halves the L2 gather traffic of the main kernel)
__global__ void h_to_f16(const float* __restrict__ h,
                         _Float16* __restrict__ hh) {
    size_t base = ((size_t)blockIdx.x * blockDim.x + threadIdx.x) * 8;
    const float4 f0 = *(const float4*)(h + base);
    const float4 f1 = *(const float4*)(h + base + 4);
    v8h hv;
    hv[0]=(_Float16)f0.x; hv[1]=(_Float16)f0.y;
    hv[2]=(_Float16)f0.z; hv[3]=(_Float16)f0.w;
    hv[4]=(_Float16)f1.x; hv[5]=(_Float16)f1.y;
    hv[6]=(_Float16)f1.z; hv[7]=(_Float16)f1.w;
    *(v8h*)(hh + base) = hv;
}

template<bool F16H>
__global__ __launch_bounds__(THREADS, 1)
void edge_mlp_kernel(const float*    __restrict__ h32,
                     const _Float16* __restrict__ h16,
                     const int*      __restrict__ src,
                     const int*      __restrict__ dst,
                     const _Float16* __restrict__ BP,
                     const float*    __restrict__ b1,
                     const float*    __restrict__ W2,
                     const float*    __restrict__ b2,
                     float*          __restrict__ out)
{
    // Double-buffered A panel: 2 x (128 rows x 40 halves, 32 used + 8 pad)
    __shared__ v8h   Ash[2][M_BLK * (ROWH / 8)];
    __shared__ float s_score[M_BLK];

    const int t    = threadIdx.x;
    const int lane = t & 31;
    const int wave = t >> 5;            // 0..15
    const int l15  = lane & 15;
    const int hi   = lane >> 4;         // lane half: 0 or 1

    const int blockStart = blockIdx.x * M_BLK;

    // ---- staging assignment: thread stages row r, 8-half segment q ----
    const int r = t >> 2;               // 0..127
    const int q = t & 3;                // 0..3
    int e  = blockStart + r;
    int ec = e < NN_EDGES ? e : (NN_EDGES - 1);
    const int si = src[ec];
    const int di = dst[ec];
    const float*    h32s = h32 + (size_t)si * HF;
    const float*    h32d = h32 + (size_t)di * HF;
    const _Float16* h16s = F16H ? (h16 + (size_t)si * HF) : nullptr;
    const _Float16* h16d = F16H ? (h16 + (size_t)di * HF) : nullptr;

    // ---- per-lane hidden columns (wave owns 32 columns = ntiles 2w, 2w+1) ----
    const int nt0 = wave * 2;
    const int nt1 = wave * 2 + 1;
    const int n0 = wave * 32 + l15;
    const int n1 = wave * 32 + 16 + l15;
    const float b1a = b1[n0], b1b = b1[n1];
    const float w2a = W2[n0], w2b = W2[n1];

    if (t < M_BLK) s_score[t] = 0.0f;

    v8f acc[8][2];
    #pragma unroll
    for (int m = 0; m < 8; ++m) { acc[m][0] = (v8f){}; acc[m][1] = (v8f){}; }

    v8h sv;                     // staged A registers (next chunk)
    auto load_stage = [&](int kb) {
        if constexpr (F16H) {
            const _Float16* rowp = (kb < HF) ? (h16s + kb) : (h16d + (kb - HF));
            sv = *(const v8h*)(rowp + q * 8);
        } else {
            const float* rowp = (kb < HF) ? (h32s + kb) : (h32d + (kb - HF));
            const float4 f0 = *(const float4*)(rowp + q * 8);
            const float4 f1 = *(const float4*)(rowp + q * 8 + 4);
            v8h hv;
            hv[0]=(_Float16)f0.x; hv[1]=(_Float16)f0.y;
            hv[2]=(_Float16)f0.z; hv[3]=(_Float16)f0.w;
            hv[4]=(_Float16)f1.x; hv[5]=(_Float16)f1.y;
            hv[6]=(_Float16)f1.z; hv[7]=(_Float16)f1.w;
            sv = hv;
        }
    };
    // B tile: contiguous 1 KB per wave from the packed layout
    auto load_B = [&](int nt, int kc) -> v16h {
        return *(const v16h*)(BP + (((size_t)nt * 32 + kc) * 32 + lane) * 16);
    };

    // ---- prologue: stage chunk 0, prefetch B for chunk 0 ----
    load_stage(0);
    Ash[0][r * (ROWH / 8) + q] = sv;
    v16h bt0 = load_B(nt0, 0);
    v16h bt1 = load_B(nt1, 0);

    int cur = 0;
    for (int kb = 0; kb < KDIM; kb += 32) {
        __syncthreads();
        const bool have_next = (kb + 32 < KDIM);
        const int  kc = kb >> 5;

        // issue next chunk's long-latency loads first (overlap with WMMAs)
        if (have_next) load_stage(kb + 32);
        v16h bt0n, bt1n;
        if (have_next) { bt0n = load_B(nt0, kc + 1); bt1n = load_B(nt1, kc + 1); }

        const _Float16* Av = (const _Float16*)Ash[cur];
        // A-fragment double buffering: load tile m+1 before WMMAs of tile m
        V16u a[2];
        {
            const int rb = l15 * ROWH;
            a[0].h[0] = *(const v8h*)(Av + rb + hi * 16);
            a[0].h[1] = *(const v8h*)(Av + rb + hi * 16 + 8);
        }
        #pragma unroll
        for (int m = 0; m < 8; ++m) {
            if (m < 7) {
                const int rb = ((m + 1) * 16 + l15) * ROWH;
                a[(m + 1) & 1].h[0] = *(const v8h*)(Av + rb + hi * 16);
                a[(m + 1) & 1].h[1] = *(const v8h*)(Av + rb + hi * 16 + 8);
            }
            acc[m][0] = __builtin_amdgcn_wmma_f32_16x16x32_f16(
                false, a[m & 1].v, false, bt0, (short)0, acc[m][0], false, false);
            acc[m][1] = __builtin_amdgcn_wmma_f32_16x16x32_f16(
                false, a[m & 1].v, false, bt1, (short)0, acc[m][1], false, false);
        }

        if (have_next) {
            Ash[cur ^ 1][r * (ROWH / 8) + q] = sv;
            bt0 = bt0n; bt1 = bt1n;
        }
        cur ^= 1;
    }

    // ---- fused layer 2: bias + ReLU + dot(W2), cross-lane + cross-wave reduce ----
    // C layout: acc[m][*][v] at lane L is hidden[m*16 + v + (L<16?0:8)][ncol(L)]
    #pragma unroll
    for (int m = 0; m < 8; ++m) {
        #pragma unroll
        for (int v = 0; v < 8; ++v) {
            float h0 = acc[m][0][v] + b1a;
            float h1 = acc[m][1][v] + b1b;
            float p = fmaxf(h0, 0.0f) * w2a + fmaxf(h1, 0.0f) * w2b;
            p += __shfl_xor(p, 1);
            p += __shfl_xor(p, 2);
            p += __shfl_xor(p, 4);
            p += __shfl_xor(p, 8);      // sum over the 16-lane N group
            if (l15 == 0) {
                atomicAdd(&s_score[m * 16 + v + hi * 8], p);  // ds_add_f32
            }
        }
    }
    __syncthreads();

    if (t < M_BLK) {
        int eo = blockStart + t;
        if (eo < NN_EDGES) out[eo] = s_score[t] + b2[0];
    }
}

extern "C" void kernel_launch(void* const* d_in, const int* in_sizes, int n_in,
                              void* d_out, int out_size, void* d_ws, size_t ws_size,
                              hipStream_t stream) {
    (void)in_sizes; (void)n_in; (void)out_size;
    const float* h   = (const float*)d_in[0];
    const int*   src = (const int*)d_in[1];
    const int*   dst = (const int*)d_in[2];
    const float* W1  = (const float*)d_in[3];
    const float* b1  = (const float*)d_in[4];
    const float* W2  = (const float*)d_in[5];
    const float* b2  = (const float*)d_in[6];
    float* out = (float*)d_out;

    const size_t HH_BYTES = (size_t)NN_NODES * HF * sizeof(_Float16);  // 51.2 MB
    const size_t BP_BYTES = (size_t)KDIM * HF * sizeof(_Float16);      // 1 MB
    const size_t BP_OFF   = (HH_BYTES + 255) & ~(size_t)255;
    const bool use_f16h = (ws_size >= BP_OFF + BP_BYTES);

    _Float16* hh = (_Float16*)d_ws;
    _Float16* BP = (_Float16*)((char*)d_ws + (use_f16h ? BP_OFF : 0));

    w1_pack_f16<<<(KDIM * HF) / 256, 256, 0, stream>>>(W1, BP);

    const int nblocks = (NN_EDGES + M_BLK - 1) / M_BLK;   // 3907 (tail clamped)
    if (use_f16h) {
        h_to_f16<<<(NN_NODES * HF / 8) / 256, 256, 0, stream>>>(h, hh);
        edge_mlp_kernel<true><<<nblocks, THREADS, 0, stream>>>(
            h, hh, src, dst, BP, b1, W2, b2, out);
    } else {
        edge_mlp_kernel<false><<<nblocks, THREADS, 0, stream>>>(
            h, nullptr, src, dst, BP, b1, W2, b2, out);
    }
}
